// SAN_79826262163505
// MI455X (gfx1250) — compile-verified, matrix operand order checked by reading, not run on hardware
//
#include <hip/hip_runtime.h>
#include <stdint.h>

typedef __attribute__((ext_vector_type(16))) _Float16 v16h;
typedef __attribute__((ext_vector_type(8)))  _Float16 v8h;
typedef __attribute__((ext_vector_type(8)))  float    v8f;

#define SAN_ALPHA 0.2f
#define SAN_NEG   -9.0e15f

// ---------------- elementwise helpers ----------------
__global__ void k_cvt_f16(const float* __restrict__ in, _Float16* __restrict__ out, int n) {
  int i = blockIdx.x * 256 + threadIdx.x;
  if (i < n) out[i] = (_Float16)in[i];
}

// in: r x c (f32, row-major) -> out: c x r (f16, row-major)  [transpose-convert]
__global__ void k_tcvt_f16(const float* __restrict__ in, _Float16* __restrict__ out, int r, int c) {
  int i = blockIdx.x * 256 + threadIdx.x;
  if (i < r * c) {
    int row = i / c, col = i - row * c;
    out[(size_t)col * r + row] = (_Float16)in[i];
  }
}

__global__ void k_zero(float* __restrict__ p, int n) {
  int i = blockIdx.x * 256 + threadIdx.x;
  if (i < n) p[i] = 0.0f;
}

__global__ void k_sigmoid(const float* __restrict__ in, float* __restrict__ out, int n) {
  int i = blockIdx.x * 256 + threadIdx.x;
  if (i < n) out[i] = 1.0f / (1.0f + __expf(-in[i]));
}

// e_row[i] = sum_c F[i][c]*att[c]; e_col[i] = sum_c F[i][c]*att[64+c]
__global__ void k_evec(const float* __restrict__ F, const float* __restrict__ att,
                       float* __restrict__ er, float* __restrict__ ec, int n) {
  int i = blockIdx.x * 256 + threadIdx.x;
  if (i >= n) return;
  float a = 0.0f, b = 0.0f;
  const float* f = F + (size_t)i * 64;
#pragma unroll 8
  for (int c = 0; c < 64; c++) {
    float v = f[c];
    a += v * att[c];
    b += v * att[64 + c];
  }
  er[i] = a; ec[i] = b;
}

// Per-row masked-softmax statistics (max, sum of exp). One wave32 per row.
__global__ void k_rowstats(const float* __restrict__ er, const float* __restrict__ ec,
                           const uint8_t* __restrict__ mask,
                           float* __restrict__ mout, float* __restrict__ zout, int n) {
  int row = blockIdx.x;
  int lane = threadIdx.x;
  const uint8_t* mp = mask + (size_t)row * n;
  float e_r = er[row];
  float m = -3.4e38f;
  for (int j = lane; j < n; j += 32) {
    float s = e_r + ec[j];
    s = (s > 0.0f) ? s : SAN_ALPHA * s;
    s = mp[j] ? s : SAN_NEG;
    m = fmaxf(m, s);
  }
#pragma unroll
  for (int o = 16; o > 0; o >>= 1) m = fmaxf(m, __shfl_xor(m, o, 32));
  float z = 0.0f;
  for (int j = lane; j < n; j += 32) {
    float s = e_r + ec[j];
    s = (s > 0.0f) ? s : SAN_ALPHA * s;
    s = mp[j] ? s : SAN_NEG;
    z += __expf(s - m);
  }
#pragma unroll
  for (int o = 16; o > 0; o >>= 1) z += __shfl_xor(z, o, 32);
  if (lane == 0) { mout[row] = m; zout[row] = z; }
}

// ---------------- generic WMMA GEMM ----------------
// C[M][N] = sum_k A[M][k] * Bt[N][k]   (A row-major MxK f16, Bt row-major NxK f16)
// block = 128 threads (4 waves); block tile 128x64; wave tile 32x64 (8 WMMA accs).
// Bt k-slab staged in LDS via global_load_async_to_lds_b128 (ASYNCcnt path).
// mode 0: Cf[M*ldc+N] = c (f32).  mode 1: Cm[M*ldc+N] = |c| > 0.5 (u8 nonzero mask).
#define LDSB_STRIDE 40
__global__ void k_gemm(const _Float16* __restrict__ A, int lda,
                       const _Float16* __restrict__ Bt, int ldb, int K,
                       float* __restrict__ Cf, uint8_t* __restrict__ Cm,
                       int ldc, int mode) {
  __shared__ _Float16 lds_b[64 * LDSB_STRIDE];
  int tid = threadIdx.x;
  int wave = tid >> 5, lane = tid & 31;
  int rb = blockIdx.x * 128 + wave * 32;  // row base (this wave, 32 rows)
  int nb = blockIdx.y * 64;               // col base (this block)
  int lrow = lane & 15;
  int kb = (lane >> 4) * 8;

  // Async staging assignment: thread covers LDS rows (tid>>2) and (tid>>2)+32,
  // 8 f16 (16B) starting at column (tid&3)*8. Fixed per thread; pointers stride 32 f16/iter.
  int srow = tid >> 2;
  int soff = (tid & 3) * 8;
  const _Float16* g0 = Bt + (size_t)(nb + srow) * ldb + soff;
  const _Float16* g1 = g0 + (size_t)32 * ldb;
  unsigned l0 = (unsigned)(uintptr_t)&lds_b[srow * LDSB_STRIDE + soff];
  unsigned l1 = (unsigned)(uintptr_t)&lds_b[(srow + 32) * LDSB_STRIDE + soff];

  // A fragment pointers for the two 16-row groups (advance 32 f16 per K-step)
  const _Float16* ap0 = A + (size_t)(rb + lrow) * lda + kb;
  const _Float16* ap1 = ap0 + (size_t)16 * lda;

  v8f acc[2][4];
#pragma unroll
  for (int g = 0; g < 2; g++)
#pragma unroll
    for (int t = 0; t < 4; t++) { v8f z = {}; acc[g][t] = z; }

  for (int k0 = 0; k0 < K; k0 += 32) {
    __syncthreads();
    // direct global -> LDS DMA (no VGPR round trip), tracked by ASYNCcnt
    asm volatile("global_load_async_to_lds_b128 %0, %1, off"
                 :: "v"(l0), "v"((unsigned long long)(uintptr_t)g0) : "memory");
    asm volatile("global_load_async_to_lds_b128 %0, %1, off"
                 :: "v"(l1), "v"((unsigned long long)(uintptr_t)g1) : "memory");
    asm volatile("s_wait_asynccnt 0" ::: "memory");
    __syncthreads();
    g0 += 32; g1 += 32;

    if (k0 + 32 < K) __builtin_prefetch(ap0 + 32, 0, 1);

    // A fragments: lane holds row, K = kb+{0..7} and kb+16+{0..7}
    v16h af[2];
#pragma unroll
    for (int g = 0; g < 2; g++) {
      const _Float16* ap = g ? ap1 : ap0;
      v8h alo = *(const v8h*)ap;
      v8h ahi = *(const v8h*)(ap + 16);
#pragma unroll
      for (int i = 0; i < 8; i++) { af[g][i] = alo[i]; af[g][8 + i] = ahi[i]; }
    }
    ap0 += 32; ap1 += 32;

#pragma unroll
    for (int t = 0; t < 4; t++) {
      const _Float16* bp = &lds_b[(t * 16 + lrow) * LDSB_STRIDE + kb];
      v8h blo = *(const v8h*)bp;
      v8h bhi = *(const v8h*)(bp + 16);
      v16h bf;
#pragma unroll
      for (int i = 0; i < 8; i++) { bf[i] = blo[i]; bf[8 + i] = bhi[i]; }
      acc[0][t] = __builtin_amdgcn_wmma_f32_16x16x32_f16(false, af[0], false, bf,
                                                         (short)0, acc[0][t], false, false);
      acc[1][t] = __builtin_amdgcn_wmma_f32_16x16x32_f16(false, af[1], false, bf,
                                                         (short)0, acc[1][t], false, false);
    }
  }

  // C layout: lane -> N = lane&15, M = r + 8*(lane>>4)
  int mtop = (lane >> 4) * 8;
#pragma unroll
  for (int g = 0; g < 2; g++) {
#pragma unroll
    for (int t = 0; t < 4; t++) {
#pragma unroll
      for (int r = 0; r < 8; r++) {
        int M = rb + g * 16 + mtop + r;
        int N = nb + t * 16 + lrow;
        float v = acc[g][t][r];
        if (mode == 0) Cf[(size_t)M * ldc + N] = v;
        else           Cm[(size_t)M * ldc + N] = (fabsf(v) > 0.5f) ? (uint8_t)1 : (uint8_t)0;
      }
    }
  }
}

// ---------------- fused masked-softmax attention x V ----------------
// O[n x 64] += softmax_row(mask ? lrelu(er_i+ec_j) : NEG) @ V, with V given as
// Vt f16 (64 x n, row-major). Probabilities built in registers as 16x32 f16 tiles.
__global__ void k_attn(const float* __restrict__ er, const float* __restrict__ ec,
                       const float* __restrict__ mrow, const float* __restrict__ zrow,
                       const uint8_t* __restrict__ mask, const _Float16* __restrict__ Vt,
                       float* __restrict__ O, int n) {
  int tid = threadIdx.x;
  int wave = tid >> 5, lane = tid & 31;
  int rb = blockIdx.x * 64 + wave * 16;
  int lrow = lane & 15;
  int kb = (lane >> 4) * 8;
  int row = rb + lrow;

  float e_r = er[row];
  float m   = mrow[row];
  float iz  = 1.0f / zrow[row];
  const uint8_t* mp = mask + (size_t)row * n;

  v8f acc[4];
#pragma unroll
  for (int t = 0; t < 4; t++) { v8f z = {}; acc[t] = z; }

  for (int k0 = 0; k0 < n; k0 += 32) {
    v16h af;
#pragma unroll
    for (int half = 0; half < 2; half++) {
      int j0 = k0 + kb + half * 16;
      uint2 mb  = *(const uint2*)(mp + j0);
      float4 e0 = *(const float4*)(ec + j0);
      float4 e1 = *(const float4*)(ec + j0 + 4);
      float ecv[8] = {e0.x, e0.y, e0.z, e0.w, e1.x, e1.y, e1.z, e1.w};
      unsigned mw[2] = {mb.x, mb.y};
#pragma unroll
      for (int t = 0; t < 8; t++) {
        float s = e_r + ecv[t];
        s = (s > 0.0f) ? s : SAN_ALPHA * s;
        bool valid = ((mw[t >> 2] >> ((t & 3) * 8)) & 0xffu) != 0u;
        s = valid ? s : SAN_NEG;
        float p = __expf(s - m) * iz;   // all-masked row: m==NEG -> p = 1/n (uniform)
        af[half * 8 + t] = (_Float16)p;
      }
    }
#pragma unroll
    for (int t = 0; t < 4; t++) {
      const _Float16* vp = &Vt[(size_t)(t * 16 + lrow) * n + k0 + kb];
      v8h blo = *(const v8h*)vp;
      v8h bhi = *(const v8h*)(vp + 16);
      v16h bf;
#pragma unroll
      for (int i = 0; i < 8; i++) { bf[i] = blo[i]; bf[8 + i] = bhi[i]; }
      acc[t] = __builtin_amdgcn_wmma_f32_16x16x32_f16(false, af, false, bf,
                                                      (short)0, acc[t], false, false);
    }
  }

  int mtop = (lane >> 4) * 8;
#pragma unroll
  for (int t = 0; t < 4; t++) {
#pragma unroll
    for (int r = 0; r < 8; r++) {
      int M = rb + mtop + r;
      int N = t * 16 + lrow;
      size_t idx = (size_t)M * 64 + N;
      O[idx] += acc[t][r];
    }
  }
}

// ---------------- host orchestration ----------------
extern "C" void kernel_launch(void* const* d_in, const int* in_sizes, int n_in,
                              void* d_out, int out_size, void* d_ws, size_t ws_size,
                              hipStream_t stream) {
  (void)in_sizes; (void)n_in; (void)out_size; (void)ws_size;
  const int NN = 2048, NE = 4096, NT = 1024;

  const float* z0 = (const float*)d_in[0];
  const float* z1 = (const float*)d_in[1];
  const float* z2 = (const float*)d_in[2];
  const float* b1 = (const float*)d_in[3];   // NN x NE
  const float* b2 = (const float*)d_in[4];   // NE x NT
  const float* Wm = (const float*)d_in[5];   // (1,64,64)
  const float* Am = (const float*)d_in[6];
  const float* Rm = (const float*)d_in[7];
  const float* Dm = (const float*)d_in[8];
  const float* att_l0_1 = (const float*)d_in[9];
  const float* att_l0_2 = (const float*)d_in[10];
  const float* att_l1_1 = (const float*)d_in[11];
  const float* att_l1_2 = (const float*)d_in[12];
  const float* att_l1_3 = (const float*)d_in[13];
  /* att_l1_4 (d_in[14]) intentionally unused: l1_4 does not affect outputs */
  const float* att_l2_1 = (const float*)d_in[15];
  const float* att_l2_2 = (const float*)d_in[16];

  char* cur = (char*)d_ws;
  auto alloc = [&](size_t bytes) -> void* {
    void* r = (void*)cur;
    cur += (bytes + 255) & ~(size_t)255;
    return r;
  };
  auto cvt = [&](const float* in, _Float16* out, int n) {
    k_cvt_f16<<<dim3((n + 255) / 256), dim3(256), 0, stream>>>(in, out, n);
  };
  auto tcvt = [&](const float* in, _Float16* out, int r, int c) {
    int n = r * c;
    k_tcvt_f16<<<dim3((n + 255) / 256), dim3(256), 0, stream>>>(in, out, r, c);
  };
  auto gemm_f = [&](const _Float16* A, int lda, const _Float16* Bt, int ldb, int K,
                    float* C, int M, int N) {
    k_gemm<<<dim3(M / 128, N / 64), dim3(128), 0, stream>>>(A, lda, Bt, ldb, K,
                                                            C, (uint8_t*)nullptr, N, 0);
  };
  auto gemm_mask = [&](const _Float16* A, int lda, int K, uint8_t* Cm, int M) {
    k_gemm<<<dim3(M / 128, M / 64), dim3(128), 0, stream>>>(A, lda, A, lda, K,
                                                            (float*)nullptr, Cm, M, 1);
  };

  // --- f16 copies of incidence matrices (entries in {-1,0,1}: exact in f16) ---
  _Float16* b1h  = (_Float16*)alloc((size_t)NN * NE * 2);  cvt(b1, b1h, NN * NE);
  _Float16* b1th = (_Float16*)alloc((size_t)NE * NN * 2);  tcvt(b1, b1th, NN, NE);
  _Float16* b2h  = (_Float16*)alloc((size_t)NE * NT * 2);  cvt(b2, b2h, NE * NT);
  _Float16* b2th = (_Float16*)alloc((size_t)NT * NE * 2);  tcvt(b2, b2th, NE, NT);

  _Float16* z0h  = (_Float16*)alloc((size_t)NN * 64 * 2);  cvt(z0, z0h, NN * 64);
  _Float16* z1h  = (_Float16*)alloc((size_t)NE * 64 * 2);  cvt(z1, z1h, NE * 64);
  _Float16* z2h  = (_Float16*)alloc((size_t)NT * 64 * 2);  cvt(z2, z2h, NT * 64);
  _Float16* z0th = (_Float16*)alloc((size_t)64 * NN * 2);  tcvt(z0, z0th, NN, 64);
  _Float16* z1th = (_Float16*)alloc((size_t)64 * NE * 2);  tcvt(z1, z1th, NE, 64);

  _Float16* Wth = (_Float16*)alloc(64 * 64 * 2);  tcvt(Wm, Wth, 64, 64);
  _Float16* Ath = (_Float16*)alloc(64 * 64 * 2);  tcvt(Am, Ath, 64, 64);
  _Float16* Rth = (_Float16*)alloc(64 * 64 * 2);  tcvt(Rm, Rth, 64, 64);
  _Float16* Dth = (_Float16*)alloc(64 * 64 * 2);  tcvt(Dm, Dth, 64, 64);

  // --- Laplacian nonzero masks via WMMA GEMM (C = A A^T style; Bt == A) ---
  uint8_t* mask0  = (uint8_t*)alloc((size_t)NN * NN);  gemm_mask(b1h,  NE, NE, mask0,  NN); // b1 b1^T
  uint8_t* mask1d = (uint8_t*)alloc((size_t)NE * NE);  gemm_mask(b1th, NN, NN, mask1d, NE); // b1^T b1
  uint8_t* mask1u = (uint8_t*)alloc((size_t)NE * NE);  gemm_mask(b2h,  NT, NT, mask1u, NE); // b2 b2^T
  uint8_t* mask2  = (uint8_t*)alloc((size_t)NT * NT);  gemm_mask(b2th, NE, NE, mask2,  NT); // b2^T b2

  // --- incidence projections ---
  float* p10 = (float*)alloc((size_t)NE * 64 * 4);  gemm_f(b1th, NN, z0th, NN, NN, p10, NE, 64); // b1^T z0
  float* p01 = (float*)alloc((size_t)NN * 64 * 4);  gemm_f(b1h,  NE, z1th, NE, NE, p01, NN, 64); // b1  z1
  float* p21 = (float*)alloc((size_t)NT * 64 * 4);  gemm_f(b2th, NE, z1th, NE, NE, p21, NT, 64); // b2^T z1
  _Float16* p10h = (_Float16*)alloc((size_t)NE * 64 * 2);  cvt(p10, p10h, NE * 64);
  _Float16* p01h = (_Float16*)alloc((size_t)NN * 64 * 2);  cvt(p01, p01h, NN * 64);
  _Float16* p21h = (_Float16*)alloc((size_t)NT * 64 * 2);  cvt(p21, p21h, NT * 64);

  // --- feature / value GEMMs: F = X @ P (f32 for e-dots) + Ft f16 (64 x n) for WMMA V ---
  auto feat = [&](const _Float16* Xh, int n, const _Float16* Pt,
                  float** Ff_out, _Float16** Ft_out) {
    float*    Ff = (float*)alloc((size_t)n * 64 * 4);
    _Float16* Ft = (_Float16*)alloc((size_t)n * 64 * 2);
    gemm_f(Xh, 64, Pt, 64, 64, Ff, n, 64);
    tcvt(Ff, Ft, n, 64);
    *Ff_out = Ff; *Ft_out = Ft;
  };
  float *F01, *F02, *F11, *F12, *F13, *F21, *F22, *V02;
  _Float16 *F01t, *F02t, *F11t, *F12t, *F13t, *F21t, *F22t, *V02t;
  feat(z0h,  NN, Wth, &F01, &F01t);   // scores+V for l0_1
  feat(z0h,  NN, Ath, &F02, &F02t);   // scores for l0_2
  feat(z1h,  NE, Wth, &F11, &F11t);   // l1_1
  feat(p10h, NE, Ath, &F12, &F12t);   // l1_2 (X = b1^T z0)
  feat(z1h,  NE, Rth, &F13, &F13t);   // l1_3
  feat(z2h,  NT, Rth, &F21, &F21t);   // l2_1
  feat(p21h, NT, Dth, &F22, &F22t);   // l2_2 (X = b2^T z1)
  feat(p01h, NN, Ath, &V02, &V02t);   // V for l0_2's powsum: (b1 z1) A

  // --- accumulators ---
  float* O0 = (float*)alloc((size_t)NN * 64 * 4);
  float* O1 = (float*)alloc((size_t)NE * 64 * 4);
  float* O2 = (float*)alloc((size_t)NT * 64 * 4);
  k_zero<<<dim3((NN * 64 + 255) / 256), dim3(256), 0, stream>>>(O0, NN * 64);
  k_zero<<<dim3((NE * 64 + 255) / 256), dim3(256), 0, stream>>>(O1, NE * 64);
  k_zero<<<dim3((NT * 64 + 255) / 256), dim3(256), 0, stream>>>(O2, NT * 64);

  // --- per-head: e-vectors, row softmax stats, fused attention x V (accumulating) ---
  auto run_head = [&](const float* Ff, const _Float16* Vt, const float* attv,
                      const uint8_t* mask, float* O, int n) {
    float* er = (float*)alloc((size_t)n * 4);
    float* ec = (float*)alloc((size_t)n * 4);
    float* mr = (float*)alloc((size_t)n * 4);
    float* zr = (float*)alloc((size_t)n * 4);
    k_evec<<<dim3((n + 255) / 256), dim3(256), 0, stream>>>(Ff, attv, er, ec, n);
    k_rowstats<<<dim3(n), dim3(32), 0, stream>>>(er, ec, mask, mr, zr, n);
    k_attn<<<dim3(n / 64), dim3(128), 0, stream>>>(er, ec, mr, zr, mask, Vt, O, n);
  };
  run_head(F01, F01t, att_l0_1, mask0,  O0, NN);
  run_head(F02, V02t, att_l0_2, mask0,  O0, NN);
  run_head(F11, F11t, att_l1_1, mask1d, O1, NE);
  run_head(F12, F12t, att_l1_2, mask1d, O1, NE);
  run_head(F13, F13t, att_l1_3, mask1u, O1, NE);
  run_head(F21, F21t, att_l2_1, mask2,  O2, NT);
  run_head(F22, F22t, att_l2_2, mask2,  O2, NT);

  // --- sigmoid into output segments ---
  float* out = (float*)d_out;
  k_sigmoid<<<dim3((NN * 64 + 255) / 256), dim3(256), 0, stream>>>(O0, out, NN * 64);
  k_sigmoid<<<dim3((NE * 64 + 255) / 256), dim3(256), 0, stream>>>(O1, out + (size_t)NN * 64, NE * 64);
  k_sigmoid<<<dim3((NT * 64 + 255) / 256), dim3(256), 0, stream>>>(O2, out + (size_t)(NN + NE) * 64, NT * 64);
}
